// HAN_47854525612559
// MI455X (gfx1250) — compile-verified, matrix operand order checked by reading, not run on hardware
//
#include <hip/hip_runtime.h>
#include <hip/hip_bf16.h>

typedef __attribute__((ext_vector_type(16))) _Float16 v16h;
typedef __attribute__((ext_vector_type(8)))  _Float16 v8h;
typedef __attribute__((ext_vector_type(8)))  float    v8f;

#define NN   20000      // nodes
#define DD   128        // feature dim
#define NE   640000     // edges per tower
#define NP   10000      // products (output cols)
#define K2   256        // 2*D
#define MB   32         // rows per block in logits kernel
#define MSUB 2          // 16-row subtiles per wave

// ---------------- helpers ----------------
__device__ __forceinline__ _Float16 hi16(float x) { return (_Float16)x; }
__device__ __forceinline__ _Float16 lo16(float x, _Float16 h) {
    return (_Float16)(x - (float)h);
}

// ---------------- gather embeddings ----------------
__global__ void __launch_bounds__(256) k_gather(const int* __restrict__ uid,
                                                const int* __restrict__ pid,
                                                const float* __restrict__ utab,
                                                const float* __restrict__ ptab,
                                                float* __restrict__ U,
                                                float* __restrict__ P) {
    int t = blockIdx.x * 256 + threadIdx.x;
    if (t >= NN * DD) return;
    int node = t >> 7, f = t & 127;
    U[t] = utab[(long long)uid[node] * DD + f];
    P[t] = ptab[(long long)pid[node] * DD + f];
}

// ---------------- degree / norm ----------------
__global__ void __launch_bounds__(256) k_deg_zero(int* du, int* dp) {
    int t = blockIdx.x * 256 + threadIdx.x;
    if (t < NN) { du[t] = 0; dp[t] = 0; }
}
__global__ void __launch_bounds__(256) k_deg_count(const int* __restrict__ eu,
                                                   const int* __restrict__ ep,
                                                   int* du, int* dp) {
    int e = blockIdx.x * 256 + threadIdx.x;
    if (e >= NE) return;
    atomicAdd(&du[eu[NE + e]], 1);   // dst of user edge
    atomicAdd(&dp[ep[NE + e]], 1);   // dst of product edge
}
__global__ void __launch_bounds__(256) k_dinv(const int* du, const int* dp,
                                              float* iu, float* ip) {
    int t = blockIdx.x * 256 + threadIdx.x;
    if (t >= NN) return;
    iu[t] = rsqrtf((float)(du[t] + 1));   // +1 self loop
    ip[t] = rsqrtf((float)(dp[t] + 1));
}

// ---------------- H = X @ W  (M=20000, K=N=128), split-f16 WMMA ----------------
__global__ void __launch_bounds__(256) k_gemm_xw(const float* __restrict__ X,
                                                 const float* __restrict__ W,
                                                 float* __restrict__ H) {
    int wave = threadIdx.x >> 5, lane = threadIdx.x & 31;
    int halfLane = lane & 15, hilo = lane >> 4;
    int tileM = blockIdx.x * 16;
    int rowA  = tileM + halfLane;     // A: lane holds row M=lane&15
    int colB  = wave * 16 + halfLane; // B: lane holds col N=lane&15
    v8f acc = {};
    for (int kc = 0; kc < 4; ++kc) {
        int kb = kc * 32 + hilo * 8;  // lanes>=16 hold K+8 per ISA layout
        v16h ah, al, bh, bl;
        const float* ap = X + (long long)rowA * DD + kb;
#pragma unroll
        for (int i = 0; i < 8; ++i) {
            float x0 = ap[i];
            float x1 = ap[i + 16];
            _Float16 h0 = hi16(x0); ah[i]     = h0; al[i]     = lo16(x0, h0);  // K = kb+i
            _Float16 h1 = hi16(x1); ah[i + 8] = h1; al[i + 8] = lo16(x1, h1);  // K = kb+i+16
            float y0 = W[(kb + i) * DD + colB];
            float y1 = W[(kb + i + 16) * DD + colB];
            _Float16 g0 = hi16(y0); bh[i]     = g0; bl[i]     = lo16(y0, g0);
            _Float16 g1 = hi16(y1); bh[i + 8] = g1; bl[i + 8] = lo16(y1, g1);
        }
        acc = __builtin_amdgcn_wmma_f32_16x16x32_f16(false, ah, false, bh, (short)0, acc, false, false);
        acc = __builtin_amdgcn_wmma_f32_16x16x32_f16(false, al, false, bh, (short)0, acc, false, false);
        acc = __builtin_amdgcn_wmma_f32_16x16x32_f16(false, ah, false, bl, (short)0, acc, false, false);
    }
#pragma unroll
    for (int i = 0; i < 8; ++i) {
        int m = hilo ? (i + 8) : i;
        H[(long long)(tileM + m) * DD + colB] = acc[i];
    }
}

// ---------------- OUT = dinv^2 * H + b  (self-loop term) ----------------
__global__ void __launch_bounds__(256) k_agg_init(const float* __restrict__ H,
                                                  const float* __restrict__ dinv,
                                                  const float* __restrict__ b,
                                                  float* __restrict__ OUT) {
    int t = blockIdx.x * 256 + threadIdx.x;
    if (t >= NN * DD) return;
    int node = t >> 7, f = t & 127;
    float di = dinv[node];
    OUT[t] = di * di * H[t] + b[f];
}

// ---------------- edge scatter: OUT[dst] += dinv[src]*dinv[dst]*H[src] ----------------
__global__ void __launch_bounds__(256) k_scatter(const int* __restrict__ ei,
                                                 const float* __restrict__ dinv,
                                                 const float* __restrict__ H,
                                                 float* __restrict__ OUT) {
    int t = blockIdx.x * 256 + threadIdx.x;   // NE*32 threads, 4 feats each
    if (t >= NE * 32) return;
    int e  = t >> 5;
    int f4 = (t & 31) * 4;
    int src = ei[e], dst = ei[NE + e];
    float norm = dinv[src] * dinv[dst];
    const float* h = H + (long long)src * DD + f4;
    float*       o = OUT + (long long)dst * DD + f4;
    float4 hv = *(const float4*)h;
    atomicAdd(&o[0], norm * hv.x);
    atomicAdd(&o[1], norm * hv.y);
    atomicAdd(&o[2], norm * hv.z);
    atomicAdd(&o[3], norm * hv.w);
}

// ---------------- out_W [256,10000] -> Wt hi/lo f16 [10000,256] ----------------
__global__ void __launch_bounds__(256) k_transpose_w(const float* __restrict__ outW,
                                                     _Float16* __restrict__ Wth,
                                                     _Float16* __restrict__ Wtl) {
    int t = blockIdx.x * 256 + threadIdx.x;
    if (t >= NP * K2) return;
    int n = t >> 8, k = t & 255;
    float x = outW[(long long)k * NP + n];
    _Float16 h = hi16(x);
    Wth[t] = h;
    Wtl[t] = lo16(x, h);
}

// ---------------- fused logits GEMM + softmax ----------------
// block = 256 threads (8 waves), owns MB=32 rows; each wave: 2 stacked 16x16 tiles
__global__ void __launch_bounds__(256) k_logits_softmax(const float* __restrict__ U,
                                                        const float* __restrict__ P,
                                                        const _Float16* __restrict__ Wth,
                                                        const _Float16* __restrict__ Wtl,
                                                        const float* __restrict__ outb,
                                                        float* __restrict__ out) {
    __shared__ __attribute__((aligned(16))) _Float16 s_hi[MB * K2];
    __shared__ __attribute__((aligned(16))) _Float16 s_lo[MB * K2];
    int tid = threadIdx.x;
    int wave = tid >> 5, lane = tid & 31, halfLane = lane & 15, hilo = lane >> 4;
    int rowBase = blockIdx.x * MB;

    // load combined = [U | P] rows into LDS with f32->f16 hi/lo split
    for (int t = tid; t < MB * K2; t += 256) {
        int r = t >> 8, c = t & 255;
        long long gr = rowBase + r;
        float x = (c < DD) ? U[gr * DD + c] : P[gr * DD + (c - DD)];
        _Float16 h = hi16(x);
        s_hi[t] = h;
        s_lo[t] = lo16(x, h);
    }
    __syncthreads();

    // phase 1: logits tiles, written to d_out
    for (int nt = wave; nt < NP / 16; nt += 8) {
        int col = nt * 16 + halfLane;
        v8f zero = {};
        v8f acc[MSUB];
#pragma unroll
        for (int s = 0; s < MSUB; ++s) acc[s] = zero;
        float bias = outb[col];
        for (int kc = 0; kc < 8; ++kc) {
            int kb = kc * 32 + hilo * 8;
            v16h bh, bl;
            const v8h* bph = (const v8h*)(Wth + (long long)col * K2 + kb);
            const v8h* bpl = (const v8h*)(Wtl + (long long)col * K2 + kb);
            v8h b0 = bph[0], b1 = bph[2], c0 = bpl[0], c1 = bpl[2];
#pragma unroll
            for (int i = 0; i < 8; ++i) { bh[i] = b0[i]; bh[i + 8] = b1[i]; bl[i] = c0[i]; bl[i + 8] = c1[i]; }
#pragma unroll
            for (int s = 0; s < MSUB; ++s) {
                int arow = s * 16 + halfLane;
                v16h ah, al;
                const v8h* aph = (const v8h*)(s_hi + arow * K2 + kb);
                const v8h* apl = (const v8h*)(s_lo + arow * K2 + kb);
                v8h a0 = aph[0], a1 = aph[2], d0 = apl[0], d1 = apl[2];
#pragma unroll
                for (int i = 0; i < 8; ++i) { ah[i] = a0[i]; ah[i + 8] = a1[i]; al[i] = d0[i]; al[i + 8] = d1[i]; }
                acc[s] = __builtin_amdgcn_wmma_f32_16x16x32_f16(false, ah, false, bh, (short)0, acc[s], false, false);
                acc[s] = __builtin_amdgcn_wmma_f32_16x16x32_f16(false, al, false, bh, (short)0, acc[s], false, false);
                acc[s] = __builtin_amdgcn_wmma_f32_16x16x32_f16(false, ah, false, bl, (short)0, acc[s], false, false);
            }
        }
#pragma unroll
        for (int s = 0; s < MSUB; ++s) {
#pragma unroll
            for (int i = 0; i < 8; ++i) {
                int m = s * 16 + (hilo ? (i + 8) : i);
                out[(long long)(rowBase + m) * NP + col] = acc[s][i] + bias;
            }
        }
    }

    __threadfence();       // make this block's logits visible to its own waves
    __syncthreads();

    // phase 2: softmax over the 32 rows this block owns (4 rows per wave)
    for (int rr = 0; rr < MB / 8; ++rr) {
        long long r = rowBase + wave * (MB / 8) + rr;
        float* rowp = out + r * NP;
        float m = -3.0e38f;
        for (int j = lane; j < NP; j += 32) m = fmaxf(m, rowp[j]);
#pragma unroll
        for (int off = 16; off > 0; off >>= 1) m = fmaxf(m, __shfl_xor(m, off, 32));
        float ssum = 0.0f;
        for (int j = lane; j < NP; j += 32) ssum += expf(rowp[j] - m);
#pragma unroll
        for (int off = 16; off > 0; off >>= 1) ssum += __shfl_xor(ssum, off, 32);
        float inv = 1.0f / ssum;
        for (int j = lane; j < NP; j += 32) rowp[j] = expf(rowp[j] - m) * inv;
    }
}

// ---------------- host orchestration ----------------
extern "C" void kernel_launch(void* const* d_in, const int* in_sizes, int n_in,
                              void* d_out, int out_size, void* d_ws, size_t ws_size,
                              hipStream_t stream) {
    (void)in_sizes; (void)n_in; (void)out_size; (void)ws_size;
    const int*   user_ids    = (const int*)d_in[0];
    const int*   product_ids = (const int*)d_in[1];
    const int*   ei_u        = (const int*)d_in[2];
    const int*   ei_p        = (const int*)d_in[3];
    const float* utab        = (const float*)d_in[4];
    const float* ptab        = (const float*)d_in[5];
    const float* uW          = (const float*)d_in[6];
    const float* ub          = (const float*)d_in[7];
    const float* pW          = (const float*)d_in[8];
    const float* pb          = (const float*)d_in[9];
    const float* outW        = (const float*)d_in[10];
    const float* outb        = (const float*)d_in[11];
    float* out = (float*)d_out;

    char* w = (char*)d_ws;
    auto alloc = [&](size_t bytes) -> char* {
        char* p = w;
        w += (bytes + 255) & ~(size_t)255;
        return p;
    };
    const size_t featB = (size_t)NN * DD * sizeof(float);
    float* Ua = (float*)alloc(featB);
    float* Ub = (float*)alloc(featB);
    float* Uc = (float*)alloc(featB);
    float* Pa = (float*)alloc(featB);
    float* Pb = (float*)alloc(featB);
    float* Pc = (float*)alloc(featB);
    float* dinv_u = (float*)alloc(NN * sizeof(float));
    float* dinv_p = (float*)alloc(NN * sizeof(float));
    int*   deg_u  = (int*)alloc(NN * sizeof(int));
    int*   deg_p  = (int*)alloc(NN * sizeof(int));
    _Float16* Wth = (_Float16*)alloc((size_t)NP * K2 * sizeof(_Float16));
    _Float16* Wtl = (_Float16*)alloc((size_t)NP * K2 * sizeof(_Float16));

    const int elemBlocks = (NN * DD + 255) / 256;      // 10000
    const int nodeBlocks = (NN + 255) / 256;           // 79
    const int edgeBlocks = (NE + 255) / 256;           // 2500
    const int scatBlocks = (NE * 32 + 255) / 256;      // 80000
    const int gemmBlocks = NN / 16;                    // 1250
    const int twBlocks   = (NP * K2 + 255) / 256;      // 10000
    const int lsBlocks   = NN / MB;                    // 625

    k_gather<<<elemBlocks, 256, 0, stream>>>(user_ids, product_ids, utab, ptab, Ua, Pa);
    k_deg_zero<<<nodeBlocks, 256, 0, stream>>>(deg_u, deg_p);
    k_deg_count<<<edgeBlocks, 256, 0, stream>>>(ei_u, ei_p, deg_u, deg_p);
    k_dinv<<<nodeBlocks, 256, 0, stream>>>(deg_u, deg_p, dinv_u, dinv_p);

    // layer 0: X=*a  H=*b  OUT=*c
    k_gemm_xw<<<gemmBlocks, 256, 0, stream>>>(Ua, uW, Ub);
    k_agg_init<<<elemBlocks, 256, 0, stream>>>(Ub, dinv_u, ub, Uc);
    k_scatter<<<scatBlocks, 256, 0, stream>>>(ei_u, dinv_u, Ub, Uc);
    k_gemm_xw<<<gemmBlocks, 256, 0, stream>>>(Pa, pW, Pb);
    k_agg_init<<<elemBlocks, 256, 0, stream>>>(Pb, dinv_p, pb, Pc);
    k_scatter<<<scatBlocks, 256, 0, stream>>>(ei_p, dinv_p, Pb, Pc);

    // layer 1: X=*c  H=*a  OUT=*b
    k_gemm_xw<<<gemmBlocks, 256, 0, stream>>>(Uc, uW + DD * DD, Ua);
    k_agg_init<<<elemBlocks, 256, 0, stream>>>(Ua, dinv_u, ub + DD, Ub);
    k_scatter<<<scatBlocks, 256, 0, stream>>>(ei_u, dinv_u, Ua, Ub);
    k_gemm_xw<<<gemmBlocks, 256, 0, stream>>>(Pc, pW + DD * DD, Pa);
    k_agg_init<<<elemBlocks, 256, 0, stream>>>(Pa, dinv_p, pb + DD, Pb);
    k_scatter<<<scatBlocks, 256, 0, stream>>>(ei_p, dinv_p, Pa, Pb);

    k_transpose_w<<<twBlocks, 256, 0, stream>>>(outW, Wth, Wtl);
    k_logits_softmax<<<lsBlocks, 256, 0, stream>>>(Ub, Pb, Wth, Wtl, outb, out);
}
